// Mhsa_47802986004933
// MI455X (gfx1250) — matrix operand order repaired and checked by reading, not compiled, not run on hardware
//
#include <hip/hip_runtime.h>

// ---------------------------------------------------------------------------
// MHSA (BN+ReLU -> QKV 1x1conv -> half attention / half conv-mix -> pool2)
// MI455X / gfx1250: everything matmul-shaped runs on v_wmma_f32_16x16x32_f16.
// wave32; A-frag = per-lane row, 2x16B loads (hi*16 / hi*16+8 K-chunks);
// B-frag = per-lane column, 1x32B load at hi*16.  f32 accumulation.
// ---------------------------------------------------------------------------

typedef _Float16 v16h __attribute__((ext_vector_type(16)));
typedef float    v8f  __attribute__((ext_vector_type(8)));

constexpr int    BATCH   = 2;
constexpr int    NTOK    = 4096;
constexpr size_t QKV_BS  = 1536ull * 4096ull;        // per-batch qkv elements

// workspace layout (bytes, all 512-aligned); total ~57.6 MB
constexpr size_t OFF_PART = 0;                        // 64*2*256 f32
constexpr size_t OFF_SB   = 131072;                   // scale/bias 2*256 f32
constexpr size_t OFF_XH   = 133632;                   // 2*4096*256 f16
constexpr size_t OFF_WF   = OFF_XH   + 4194304;       // 1536*256 f16
constexpr size_t OFF_W1   = OFF_WF   + 786432;        // 256*768 f16
constexpr size_t OFF_W2   = OFF_W1   + 393216;        // 256*768 f16
constexpr size_t OFF_QKVH = OFF_W2   + 393216;        // 2*1536*4096 f16
constexpr size_t OFF_VT   = OFF_QKVH + 25165824;      // 2*4*64*4096 f16
constexpr size_t OFF_QCOL = OFF_VT   + 4194304;       // 2*4096*768 f16
constexpr size_t OFF_KCOL = OFF_QCOL + 12582912;      // 2*4096*768 f16

union FragH { v16h h; uint4 u[2]; };

__device__ __forceinline__ v16h loadA(const _Float16* row, int hi) {
  // A 16x32, k order chosen here (matches loadB): lanes<16 hold K 0..15 ; lanes>=16 hold 16..31
  FragH f;
  f.u[0] = *reinterpret_cast<const uint4*>(row + hi * 16);
  f.u[1] = *reinterpret_cast<const uint4*>(row + hi * 16 + 8);
  return f.h;
}
__device__ __forceinline__ v16h loadB(const _Float16* col, int hi) {
  // B 32x16, k order chosen here (matches loadA): lane = column, lanes<16 hold K 0..15, lanes>=16 hold 16..31
  FragH f;
  f.u[0] = *reinterpret_cast<const uint4*>(col + hi * 16);
  f.u[1] = *reinterpret_cast<const uint4*>(col + hi * 16 + 8);
  return f.h;
}
__device__ __forceinline__ v8f wmma16(v16h a, v16h b, v8f c) {
  return __builtin_amdgcn_wmma_f32_16x16x32_f16(false, a, false, b, (short)0, c,
                                                false, false);
}
__device__ __forceinline__ unsigned pack2h(float a, float b) {
  union { _Float16 h[2]; unsigned u; } t;
  t.h[0] = (_Float16)a; t.h[1] = (_Float16)b;
  return t.u;
}
__device__ __forceinline__ v8f zero8() { v8f z = {0,0,0,0,0,0,0,0}; return z; }

// ------------------------------ BatchNorm stats -----------------------------
__global__ __launch_bounds__(256) void bn_partial(const float* __restrict__ x,
                                                  float* __restrict__ part) {
  int j = blockIdx.x, t = threadIdx.x;   // 64 blocks x 128 rows, channel = t
  float s = 0.f, q = 0.f;
  for (int r = j * 128; r < j * 128 + 128; ++r) {
    float v = x[(size_t)r * 256 + t];
    s += v; q += v * v;
  }
  part[(size_t)(j * 2 + 0) * 256 + t] = s;
  part[(size_t)(j * 2 + 1) * 256 + t] = q;
}
__global__ __launch_bounds__(256) void bn_final(const float* __restrict__ part,
                                                const float* __restrict__ gamma,
                                                const float* __restrict__ beta,
                                                float* __restrict__ sb) {
  int t = threadIdx.x;
  float s = 0.f, q = 0.f;
  for (int j = 0; j < 64; ++j) {       // fixed order -> deterministic
    s += part[(size_t)(j * 2 + 0) * 256 + t];
    q += part[(size_t)(j * 2 + 1) * 256 + t];
  }
  float mean = s * (1.f / 8192.f);
  float var  = q * (1.f / 8192.f) - mean * mean;   // biased var, as reference
  float rs   = rsqrtf(var + 1e-5f);
  float sc   = gamma[t] * rs;
  sb[t] = sc; sb[256 + t] = beta[t] - mean * sc;
}

// --------------------- fused BN-apply + ReLU + f16 convert ------------------
__global__ __launch_bounds__(256) void xh_kernel(const float* __restrict__ x,
                                                 const float* __restrict__ sb,
                                                 _Float16* __restrict__ xh) {
  int i = blockIdx.x * 256 + threadIdx.x;          // 2*4096*256 threads
  int c = i & 255;
  float v = fmaf(x[i], sb[c], sb[256 + c]);
  xh[i] = (_Float16)fmaxf(v, 0.f);
}

// ----------------------------- weight f16 packs -----------------------------
__global__ __launch_bounds__(256) void pack_w(const float* __restrict__ qkv_w,
                                              const float* __restrict__ c1,
                                              const float* __restrict__ c2,
                                              _Float16* __restrict__ wf,
                                              _Float16* __restrict__ w1h,
                                              _Float16* __restrict__ w2h) {
  int i = blockIdx.x * 256 + threadIdx.x;          // 786432 threads
  if (i < 393216)       wf[i] = (_Float16)qkv_w[i];
  else if (i < 589824)  w1h[i - 393216] = (_Float16)c1[i - 393216];
  else                  w2h[i - 589824] = (_Float16)c2[i - 589824];
}

// ------------------------ QKV GEMM: qkv[o,s] = W @ Xn -----------------------
// per wave: 64(M) x 32(N) tile, K=256 ; M=1536, N=4096 per batch
__global__ __launch_bounds__(256) void qkv_gemm(const _Float16* __restrict__ xh,
                                                const _Float16* __restrict__ wf,
                                                _Float16* __restrict__ qkvh) {
  int lane = threadIdx.x & 31, hi = lane >> 4, lo = lane & 15;
  int wid = blockIdx.x * 8 + (threadIdx.x >> 5);   // 6144 waves
  int b = wid / 3072, r = wid % 3072;
  int m0 = (r >> 7) * 64, n0 = (r & 127) * 32;
  const _Float16* xb = xh + (size_t)b * (4096ull * 256ull);

  v8f acc[4][2];
  #pragma unroll
  for (int tm = 0; tm < 4; ++tm)
    #pragma unroll
    for (int tn = 0; tn < 2; ++tn) acc[tm][tn] = zero8();

  for (int kk = 0; kk < 256; kk += 32) {
    v16h Bf[2];
    #pragma unroll
    for (int tn = 0; tn < 2; ++tn)
      Bf[tn] = loadB(xb + (size_t)(n0 + tn * 16 + lo) * 256 + kk, hi);
    #pragma unroll
    for (int tm = 0; tm < 4; ++tm) {
      v16h Af = loadA(wf + (size_t)(m0 + tm * 16 + lo) * 256 + kk, hi);
      #pragma unroll
      for (int tn = 0; tn < 2; ++tn)
        acc[tm][tn] = wmma16(Af, Bf[tn], acc[tm][tn]);
    }
  }
  _Float16* ob = qkvh + (size_t)b * QKV_BS;
  #pragma unroll
  for (int tm = 0; tm < 4; ++tm)
    #pragma unroll
    for (int tn = 0; tn < 2; ++tn) {
      int s = n0 + tn * 16 + lo;
      #pragma unroll
      for (int rr = 0; rr < 8; ++rr) {
        int o = m0 + tm * 16 + rr + 8 * hi;
        ob[(size_t)o * 4096 + s] = (_Float16)acc[tm][tn][rr];
      }
    }
}

// --------------------- V^T transpose for attention heads 0..3 ---------------
__global__ __launch_bounds__(256) void vt_kernel(const _Float16* __restrict__ qkvh,
                                                 _Float16* __restrict__ vt) {
  __shared__ _Float16 tile[64][72];
  int bid = blockIdx.x;                 // 512 = 2b * 4h * 64 n-chunks
  int b = bid >> 8, h = (bid >> 6) & 3, n0 = (bid & 63) * 64;
  int t = threadIdx.x;
  int nl = t >> 2, dq = (t & 3) * 16;
  const _Float16* src = qkvh + (size_t)b * QKV_BS +
                        (size_t)(n0 + nl) * 1536 + 1024 + h * 64 + dq;
  *reinterpret_cast<uint4*>(&tile[nl][dq])     = *reinterpret_cast<const uint4*>(src);
  *reinterpret_cast<uint4*>(&tile[nl][dq + 8]) = *reinterpret_cast<const uint4*>(src + 8);
  __syncthreads();
  int d = t >> 2, nq = (t & 3) * 16;
  alignas(16) _Float16 tmp[16];
  #pragma unroll
  for (int j = 0; j < 16; ++j) tmp[j] = tile[nq + j][d];
  _Float16* dst = vt + ((size_t)((b * 4 + h) * 64 + d)) * 4096 + n0 + nq;
  *reinterpret_cast<uint4*>(dst)     = *reinterpret_cast<uint4*>(&tmp[0]);
  *reinterpret_cast<uint4*>(dst + 8) = *reinterpret_cast<uint4*>(&tmp[8]);
}

// ---------------- im2col for conv-mix (heads 4..7 of q and k) ---------------
// qcol/kcol: [b][s=y*64+x][c*3+t] f16, zero padded at image edges.
__global__ __launch_bounds__(256) void im2col_kernel(const _Float16* __restrict__ qkvh,
                                                     _Float16* __restrict__ qcol,
                                                     _Float16* __restrict__ kcol) {
  int idx = blockIdx.x * 256 + threadIdx.x;        // 32768 = 2b*256c*64y
  int b = idx >> 14, rem = idx & 16383;
  int c = rem >> 6, y = rem & 63;
  int hsel = c >> 6;
  int n2 = (c & 63) * 64 + y;
  const _Float16* base = qkvh + (size_t)b * QKV_BS + (size_t)n2 * 1536;
  const _Float16* qr = base + 256 + hsel * 64;     // q2[c,y,:]
  const _Float16* kr = base + 768 + hsel * 64;     // k2[c,y,:]
  _Float16* qo = qcol + ((size_t)(b * 4096 + y * 64)) * 768 + c * 3;
  _Float16* ko = kcol + ((size_t)(b * 4096 + y * 64)) * 768 + c * 3;
  bool ym = (y > 0), yp = (y < 63);
  const _Float16 Z = (_Float16)0.f;
  for (int x = 0; x < 64; ++x) {
    size_t off = (size_t)x * 768;
    qo[off + 0] = ym ? qr[x - 1536] : Z;   // tap y-1
    qo[off + 1] = qr[x];                   // tap y
    qo[off + 2] = yp ? qr[x + 1536] : Z;   // tap y+1
    ko[off + 0] = (x > 0)  ? kr[x - 1] : Z;
    ko[off + 1] = kr[x];
    ko[off + 2] = (x < 63) ? kr[x + 1] : Z;
  }
}

// ------------------- flash attention (heads 0..3), WMMA ---------------------
// per wave: one 16-query tile of one (b,h); S^T = K*Q^T ; O^T = V^T * P^T.
__global__ __launch_bounds__(256) void attn_kernel(const _Float16* __restrict__ qkvh,
                                                   const _Float16* __restrict__ vt,
                                                   float* __restrict__ out) {
  const float SC = 0.180336880111120f;   // (1/8) * log2(e)
  int lane = threadIdx.x & 31, hi = lane >> 4, lo = lane & 15;
  int wid = blockIdx.x * 8 + (threadIdx.x >> 5);   // 2048 waves
  int qt = wid & 255, h = (wid >> 8) & 3, b = wid >> 10;
  int q0 = qt * 16;
  const _Float16* qb = qkvh + (size_t)b * QKV_BS;
  const _Float16* qrow = qb + (size_t)(q0 + lo) * 1536 + h * 64;
  v16h Bq0 = loadB(qrow, hi);            // Q^T cols, c 0..31
  v16h Bq1 = loadB(qrow + 32, hi);       // c 32..63
  const _Float16* vtb = vt + (size_t)((b * 4 + h) * 64) * 4096;

  v8f acc[4]; 
  #pragma unroll
  for (int t = 0; t < 4; ++t) acc[t] = zero8();
  float m = -1e30f, l = 0.f;

  for (int kb = 0; kb < 4096; kb += 32) {
    const _Float16* k0r = qb + (size_t)(kb + lo) * 1536 + 512 + h * 64;
    const _Float16* k1r = qb + (size_t)(kb + 16 + lo) * 1536 + 512 + h * 64;
    v8f st0 = zero8(), st1 = zero8();
    st0 = wmma16(loadA(k0r, hi),      Bq0, st0);
    st0 = wmma16(loadA(k0r + 32, hi), Bq1, st0);
    st1 = wmma16(loadA(k1r, hi),      Bq0, st1);
    st1 = wmma16(loadA(k1r + 32, hi), Bq1, st1);

    float mloc = st0[0];
    #pragma unroll
    for (int r = 0; r < 8; ++r) { mloc = fmaxf(mloc, st0[r]); mloc = fmaxf(mloc, st1[r]); }
    mloc = fmaxf(mloc, __shfl_xor(mloc, 16, 32));
    float mnew  = fmaxf(m, mloc);
    float alpha = __builtin_amdgcn_exp2f((m - mnew) * SC);

    float p0[8], p1[8], ls = 0.f;
    #pragma unroll
    for (int r = 0; r < 8; ++r) {
      p0[r] = __builtin_amdgcn_exp2f((st0[r] - mnew) * SC);
      p1[r] = __builtin_amdgcn_exp2f((st1[r] - mnew) * SC);
      ls += p0[r] + p1[r];
    }
    ls += __shfl_xor(ls, 16, 32);
    l = l * alpha + ls;

    unsigned d0[4], d1[4], o0[4], o1[4];
    #pragma unroll
    for (int i = 0; i < 4; ++i) {
      d0[i] = pack2h(p0[2 * i], p0[2 * i + 1]);
      d1[i] = pack2h(p1[2 * i], p1[2 * i + 1]);
    }
    #pragma unroll
    for (int i = 0; i < 4; ++i) {
      o0[i] = (unsigned)__shfl_xor((int)d0[i], 16, 32);
      o1[i] = (unsigned)__shfl_xor((int)d1[i], 16, 32);
    }
    union { v16h h; unsigned u[8]; } bp;   // P^T as B-frag (32k x 16q)
    #pragma unroll
    for (int i = 0; i < 4; ++i) {
      bp.u[i]     = hi ? o1[i] : d0[i];
      bp.u[4 + i] = hi ? d1[i] : o0[i];
    }
    #pragma unroll
    for (int t = 0; t < 4; ++t) {
      acc[t] *= alpha;
      v16h av = loadA(vtb + (size_t)(t * 16 + lo) * 4096 + kb, hi);  // V^T rows
      acc[t] = wmma16(av, bp.h, acc[t]);
    }
    m = mnew;
  }

  // epilogue: O^T[d,q] per lane (col q, 8 consecutive d) -> pair-avg -> out[:,0:128)
  float rl = 0.5f / l;
  float* orow = out + ((size_t)(b * 4096 + q0 + lo)) * 256 + h * 32;
  #pragma unroll
  for (int t = 0; t < 4; ++t) {
    float4 res;
    res.x = (acc[t][0] + acc[t][1]) * rl;
    res.y = (acc[t][2] + acc[t][3]) * rl;
    res.z = (acc[t][4] + acc[t][5]) * rl;
    res.w = (acc[t][6] + acc[t][7]) * rl;
    *reinterpret_cast<float4*>(orow + 8 * t + 4 * hi) = res;
  }
}

// ------------- conv-mix GEMMs (K=768 x2) + v2 add + pooled store ------------
__global__ __launch_bounds__(256) void conv_gemm(const _Float16* __restrict__ qcol,
                                                 const _Float16* __restrict__ kcol,
                                                 const _Float16* __restrict__ w1h,
                                                 const _Float16* __restrict__ w2h,
                                                 const _Float16* __restrict__ qkvh,
                                                 float* __restrict__ out) {
  int lane = threadIdx.x & 31, hi = lane >> 4, lo = lane & 15;
  int wid = blockIdx.x * 8 + (threadIdx.x >> 5);   // 1024 waves
  int b = wid >> 9, r = wid & 511;
  int m0 = (r >> 7) * 64, s0 = (r & 127) * 32;

  v8f acc[4][2];
  #pragma unroll
  for (int tm = 0; tm < 4; ++tm)
    #pragma unroll
    for (int tn = 0; tn < 2; ++tn) acc[tm][tn] = zero8();

  for (int pass = 0; pass < 2; ++pass) {
    const _Float16* col = (pass ? kcol : qcol) + (size_t)b * (4096ull * 768ull);
    const _Float16* wgt = pass ? w2h : w1h;
    for (int kk = 0; kk < 768; kk += 32) {
      v16h Bf[2];
      #pragma unroll
      for (int tn = 0; tn < 2; ++tn)
        Bf[tn] = loadB(col + (size_t)(s0 + tn * 16 + lo) * 768 + kk, hi);
      #pragma unroll
      for (int tm = 0; tm < 4; ++tm) {
        v16h Af = loadA(wgt + (size_t)(m0 + tm * 16 + lo) * 768 + kk, hi);
        #pragma unroll
        for (int tn = 0; tn < 2; ++tn)
          acc[tm][tn] = wmma16(Af, Bf[tn], acc[tm][tn]);
      }
    }
  }

  const _Float16* qb = qkvh + (size_t)b * QKV_BS;
  #pragma unroll
  for (int tm = 0; tm < 4; ++tm)
    #pragma unroll
    for (int tn = 0; tn < 2; ++tn) {
      int s = s0 + tn * 16 + lo;
      int y = s >> 6, x = s & 63;
      #pragma unroll
      for (int rr = 0; rr < 8; ++rr) {
        int o  = m0 + tm * 16 + rr + 8 * hi;
        int n2 = (o & 63) * 64 + y;        // v2[b,o,y,x] via raw view
        float val = acc[tm][tn][rr] +
                    (float)qb[(size_t)n2 * 1536 + 1280 + (o >> 6) * 64 + x];
        float par = __shfl_xor(val, 1, 32);
        if ((lane & 1) == 0) {             // x even: average (x, x+1)
          int n = (o << 4) | (y >> 2);
          int c = 128 + (y & 3) * 32 + (x >> 1);
          out[((size_t)b * 4096 + n) * 256 + c] = 0.5f * (val + par);
        }
      }
    }
}

// ---------------------------------------------------------------------------
extern "C" void kernel_launch(void* const* d_in, const int* in_sizes, int n_in,
                              void* d_out, int out_size, void* d_ws, size_t ws_size,
                              hipStream_t stream) {
  (void)in_sizes; (void)n_in; (void)out_size; (void)ws_size;
  const float* x     = (const float*)d_in[0];
  const float* qkv_w = (const float*)d_in[1];
  const float* gamma = (const float*)d_in[2];
  const float* beta  = (const float*)d_in[3];
  const float* c1w   = (const float*)d_in[4];
  const float* c2w   = (const float*)d_in[5];
  float* out = (float*)d_out;

  char* ws = (char*)d_ws;
  float*    part = (float*)   (ws + OFF_PART);
  float*    sb   = (float*)   (ws + OFF_SB);
  _Float16* xh   = (_Float16*)(ws + OFF_XH);
  _Float16* wf   = (_Float16*)(ws + OFF_WF);
  _Float16* w1h  = (_Float16*)(ws + OFF_W1);
  _Float16* w2h  = (_Float16*)(ws + OFF_W2);
  _Float16* qkvh = (_Float16*)(ws + OFF_QKVH);
  _Float16* vt   = (_Float16*)(ws + OFF_VT);
  _Float16* qcol = (_Float16*)(ws + OFF_QCOL);
  _Float16* kcol = (_Float16*)(ws + OFF_KCOL);

  bn_partial   <<<  64, 256, 0, stream>>>(x, part);
  bn_final     <<<   1, 256, 0, stream>>>(part, gamma, beta, sb);
  xh_kernel    <<<8192, 256, 0, stream>>>(x, sb, xh);
  pack_w       <<<3072, 256, 0, stream>>>(qkv_w, c1w, c2w, wf, w1h, w2h);
  qkv_gemm     <<< 768, 256, 0, stream>>>(xh, wf, qkvh);
  vt_kernel    <<< 512, 256, 0, stream>>>(qkvh, vt);
  im2col_kernel<<< 128, 256, 0, stream>>>(qkvh, qcol, kcol);
  attn_kernel  <<< 256, 256, 0, stream>>>(qkvh, vt, out);
  conv_gemm    <<< 128, 256, 0, stream>>>(qcol, kcol, w1h, w2h, qkvh, out);
}
